// PointCloud3DCNN_41669772706562
// MI455X (gfx1250) — matrix operand order repaired and checked by reading, not compiled, not run on hardware
//
#include <hip/hip_runtime.h>
#include <stdint.h>

typedef __attribute__((ext_vector_type(16))) _Float16 v16h;
typedef __attribute__((ext_vector_type(8)))  _Float16 v8h;
typedef __attribute__((ext_vector_type(8)))  float    v8f;

#define DV 96
#define HV 96
#define WVX 48

// ---------------------------------------------------------------------------
// Pack fp32 OIDHW weights -> f16 [Co][Kpad], K = tap*Cipad + ci (tap-major),
// zero-padded in ci (Ci->Cipad) and K tail (Ktot->Kpad, Kpad % 32 == 0).
// Folds conv-bias into BN affine: y = acc*sc[c] + sh[c]; sc=s, sh=b*s+t.
// ---------------------------------------------------------------------------
__global__ void k_pack_weights(const float* __restrict__ w, const float* __restrict__ b,
                               const float* __restrict__ s, const float* __restrict__ t,
                               _Float16* __restrict__ Wp, float* __restrict__ sc,
                               float* __restrict__ sh, int Co, int Ci, int Cipad, int Kpad)
{
    int i = blockIdx.x * blockDim.x + threadIdx.x;
    int total = Co * Kpad;
    if (i >= total) return;
    int co = i / Kpad, k = i - co * Kpad;
    int tap = k / Cipad, ci = k - tap * Cipad;
    float v = 0.f;
    if (tap < 27 && ci < Ci) v = w[(co * Ci + ci) * 27 + tap];   // OIDHW, (kd,kh,kw)->tap
    Wp[i] = (_Float16)v;
    if (k == 0) { float ss = s[co]; sc[co] = ss; sh[co] = b[co] * ss + t[co]; }
}

// ---------------------------------------------------------------------------
// Scatter (B,N,9) point features into dense channels-last f16 [pos][16] + mask
// ---------------------------------------------------------------------------
__global__ void k_densify(const float* __restrict__ vf, const int* __restrict__ coords,
                          _Float16* __restrict__ X, float* __restrict__ m, int B, int N)
{
    int i = blockIdx.x * blockDim.x + threadIdx.x;
    if (i >= B * N) return;
    int cd = coords[3 * i], ch = coords[3 * i + 1], cw = coords[3 * i + 2];
    int b = i / N;
    long pos = (((long)b * DV + cd) * HV + ch) * (long)WVX + cw;
    #pragma unroll
    for (int c = 0; c < 9; ++c) X[pos * 16 + c] = (_Float16)vf[(long)i * 9 + c];
    m[pos] = 1.f;
}

// ---------------------------------------------------------------------------
// Active-set downsample: any active input in 3^3 stride-2 VALID window
// ---------------------------------------------------------------------------
__global__ void k_down_mask(const float* __restrict__ mi, float* __restrict__ mo,
                            int B, int Di, int Hi, int Wi, int Do, int Ho, int Wo)
{
    int i = blockIdx.x * blockDim.x + threadIdx.x;
    int spat = Do * Ho * Wo;
    if (i >= B * spat) return;
    int b = i / spat, r = i - b * spat;
    int od = r / (Ho * Wo); r -= od * (Ho * Wo);
    int oh = r / Wo, ow = r - (r / Wo) * Wo;
    float any = 0.f;
    for (int kd = 0; kd < 3; ++kd)
        for (int kh = 0; kh < 3; ++kh)
            for (int kw = 0; kw < 3; ++kw) {
                int id = 2 * od + kd, ih = 2 * oh + kh, iw = 2 * ow + kw;
                if (id < Di && ih < Hi && iw < Wi)
                    any += mi[(((long)b * Di + id) * Hi + ih) * (long)Wi + iw];
            }
    mo[i] = any > 0.f ? 1.f : 0.f;
}

// ---------------------------------------------------------------------------
// Implicit-GEMM 3^3 conv, WMMA f32<-f16, templated on channel count / mode /
// N-tiling.  Block = 8 waves; each wave owns one 16-row M tile; all waves in
// the block share one (NT*16)-wide co-block whose weight chunks are streamed
// into double-buffered LDS with GLOBAL_LOAD_ASYNC_TO_LDS_B64 (ASYNCcnt),
// one s_wait_asynccnt + __syncthreads per 32-deep K chunk.
// A-fragment per ISA 16-bit layout: lane holds K runs [half*8..+7] and
// [16+half*8..+7]  (two 16B loads; two taps when CIPAD==16).
// MODE: 0 = stride1 SAME (subm), 1 = stride2 VALID (down),
//       2 = transposed conv (lhs_dilation 2, pad 2, tail-pad semantics).
// Epilogue fuses BN affine + ReLU + active-mask, f16 channels-last store.
// ---------------------------------------------------------------------------
template<int CIPAD, int MODE, int NT>
__global__ __launch_bounds__(256)
void k_conv_wmma(const _Float16* __restrict__ X, const _Float16* __restrict__ Wp,
                 const float* __restrict__ sc, const float* __restrict__ sh,
                 const float* __restrict__ mask, _Float16* __restrict__ Y,
                 int B, int Di, int Hi, int Wi, int Do, int Ho, int Wo, int Co)
{
    constexpr int KPAD  = ((27 * CIPAD + 31) / 32) * 32;
    constexpr int NIT   = KPAD / 32;
    constexpr int NCOLS = NT * 16;
    constexpr int LSTR  = 40;                 // padded LDS column stride (halfs, 80B)

    __shared__ __align__(16) _Float16 Bs[2][NCOLS * LSTR];

    const int tid  = threadIdx.x;
    const int lane = tid & 31;
    const int wv   = tid >> 5;
    const int half = lane >> 4;
    const int l16  = lane & 15;

    const int spat   = Do * Ho * Wo;
    const int totalM = B * spat;
    const int mTile  = blockIdx.x * 8 + wv;
    const int coBase = blockIdx.y * NCOLS;

    // A-fragment row (output position) for this lane
    int rowA = (mTile << 4) + l16;
    const bool rowValid = rowA < totalM;
    int rr = rowValid ? rowA : 0;
    int b  = rr / spat; int r2 = rr - b * spat;
    int od = r2 / (Ho * Wo); r2 -= od * (Ho * Wo);
    int oh = r2 / Wo; int ow = r2 - oh * Wo;

    // async cooperative weight-chunk staging: NCOLS cols x 32 K halfs -> LDS
    auto stage = [&](int it, int bsel) {
        const int k0 = it * 32;
        for (int u = tid; u < NCOLS * 8; u += 256) {
            int col = u >> 3, j4 = (u & 7) << 2;
            uint32_t lds = (uint32_t)(uintptr_t)&Bs[bsel][col * LSTR + j4];
            uint64_t ga  = (uint64_t)(uintptr_t)(Wp + (size_t)(coBase + col) * KPAD + k0 + j4);
            asm volatile("global_load_async_to_lds_b64 %0, %1, off"
                         :: "v"(lds), "v"(ga) : "memory");
        }
    };
    auto wait_async = []() { asm volatile("s_wait_asynccnt 0x0" ::: "memory"); };

    // per-tap input pointer (nullptr -> contribute zeros)
    auto aptr = [&](int tap) -> const _Float16* {
        if (!rowValid || tap >= 27) return (const _Float16*)nullptr;
        const int kd = tap / 9;
        const int rem = tap - kd * 9;
        const int kh = rem / 3, kw = rem - (rem / 3) * 3;
        int id, ih, iw; bool ok = true;
        if constexpr (MODE == 0) { id = od + kd - 1; ih = oh + kh - 1; iw = ow + kw - 1; }
        else if constexpr (MODE == 1) { id = 2 * od + kd; ih = 2 * oh + kh; iw = 2 * ow + kw; }
        else {                                 // transposed: parity + range
            int td = od + kd - 2, th = oh + kh - 2, tw = ow + kw - 2;
            ok = (((td | th | tw) & 1) == 0);
            id = td >> 1; ih = th >> 1; iw = tw >> 1;
        }
        ok = ok && (unsigned)id < (unsigned)Di && (unsigned)ih < (unsigned)Hi
                && (unsigned)iw < (unsigned)Wi;
        if (!ok) return (const _Float16*)nullptr;
        long pin = (((long)b * Di + id) * Hi + ih) * (long)Wi + iw;
        return X + pin * CIPAD;
    };

    stage(0, 0);

    v8f acc[NT] = {};
    int buf = 0;
    for (int it = 0; it < NIT; ++it) {
        wait_async();                          // my async stores to LDS landed
        __syncthreads();                       // everyone's staging of `buf` done
        if (it + 1 < NIT) stage(it + 1, buf ^ 1);

        // ---- A fragment (ISA layout: K runs half*8..+7 and 16+half*8..+7) ----
        v8h alo = {}, ahi = {};
        if constexpr (CIPAD >= 32) {
            const int tap = (it * 32) / CIPAD;         // whole chunk in one tap
            const int ciB = (it * 32 & (CIPAD - 1)) + (half << 3);
            const _Float16* p = aptr(tap);
            if (p) { alo = *(const v8h*)(p + ciB); ahi = *(const v8h*)(p + ciB + 16); }
        } else {                                       // CIPAD==16: two taps
            const _Float16* p0 = aptr(2 * it);
            const _Float16* p1 = aptr(2 * it + 1);
            const int ciB = half << 3;
            if (p0) alo = *(const v8h*)(p0 + ciB);
            if (p1) ahi = *(const v8h*)(p1 + ciB);
        }
        v16h a = __builtin_shufflevector(alo, ahi,
                                         0, 1, 2, 3, 4, 5, 6, 7,
                                         8, 9, 10, 11, 12, 13, 14, 15);

        #pragma unroll
        for (int n = 0; n < NT; ++n) {
            // B fragment: lane group holds K runs half*8..+7 and 16+half*8..+7 of col
            const _Float16* bp = &Bs[buf][(n * 16 + l16) * LSTR + (half << 3)];
            v8h blo = *(const v8h*)bp, bhi = *(const v8h*)(bp + 16);
            v16h bf = __builtin_shufflevector(blo, bhi, 0, 1, 2, 3, 4, 5, 6, 7,
                                              8, 9, 10, 11, 12, 13, 14, 15);
            acc[n] = __builtin_amdgcn_wmma_f32_16x16x32_f16(
                         false, a, false, bf, (short)0, acc[n], false, false);
        }
        buf ^= 1;
    }

    // Fused BN + ReLU + active-set mask epilogue; f16 channels-last store.
    #pragma unroll
    for (int n = 0; n < NT; ++n) {
        const int co = coBase + n * 16 + l16;
        const float s = sc[co], t = sh[co];
        #pragma unroll
        for (int r = 0; r < 8; ++r) {
            int row = (mTile << 4) + r + (half << 3);    // C/D layout: M = r + 8*half
            if (row < totalM) {
                float v = acc[n][r] * s + t;
                v = v > 0.f ? v : 0.f;
                v *= mask[row];
                Y[(long)row * Co + co] = (_Float16)v;
            }
        }
    }
}

static void run_conv(int Cipad, int mode, int NT,
                     const _Float16* X, const _Float16* Wp, const float* sc, const float* sh,
                     const float* mask, _Float16* Y, int B,
                     int Di, int Hi, int Wi, int Do, int Ho, int Wo, int Co,
                     hipStream_t stream)
{
    int totalM = B * Do * Ho * Wo;
    int tilesM = (totalM + 15) / 16;
    dim3 grid((tilesM + 7) / 8, Co / (NT * 16));
#define CONV_CASE(CP, MD, NTV)                                                   \
    if (Cipad == CP && mode == MD && NT == NTV) {                                \
        k_conv_wmma<CP, MD, NTV><<<grid, 256, 0, stream>>>(                      \
            X, Wp, sc, sh, mask, Y, B, Di, Hi, Wi, Do, Ho, Wo, Co);              \
        return;                                                                  \
    }
    CONV_CASE(16, 0, 1)  CONV_CASE(16, 1, 2)
    CONV_CASE(32, 0, 2)  CONV_CASE(32, 1, 4)  CONV_CASE(32, 2, 1)
    CONV_CASE(64, 0, 4)  CONV_CASE(64, 1, 4)  CONV_CASE(64, 2, 2)
    CONV_CASE(128, 0, 4) CONV_CASE(128, 1, 4) CONV_CASE(128, 2, 4)
    CONV_CASE(256, 0, 4) CONV_CASE(256, 2, 4)
#undef CONV_CASE
}

// ---------------------------------------------------------------------------
// 1x1x1 classifier head: 3 logits, softmax, class-1 prob * mask -> fp32 out
// ---------------------------------------------------------------------------
__global__ void k_cls(const _Float16* __restrict__ A, const float* __restrict__ w,
                      const float* __restrict__ bb, const float* __restrict__ mask,
                      float* __restrict__ out, int total, int Ci, int Cpad)
{
    int i = blockIdx.x * blockDim.x + threadIdx.x;
    if (i >= total) return;
    float z0 = bb[0], z1 = bb[1], z2 = bb[2];
    const _Float16* a = A + (long)i * Cpad;
    for (int c = 0; c < Ci; ++c) {
        float x = (float)a[c];
        z0 += w[c] * x; z1 += w[Ci + c] * x; z2 += w[2 * Ci + c] * x;
    }
    float mx = fmaxf(z0, fmaxf(z1, z2));
    float e0 = __expf(z0 - mx), e1 = __expf(z1 - mx), e2 = __expf(z2 - mx);
    out[i] = (e1 / (e0 + e1 + e2)) * mask[i];
}

__global__ void k_add(_Float16* __restrict__ d, const _Float16* __restrict__ x, long n)
{
    long i = (long)blockIdx.x * blockDim.x + threadIdx.x;
    if (i < n) d[i] = (_Float16)((float)d[i] + (float)x[i]);
}

// occ head: Conv3d(16,1,k3,pad1) on dense d0, fp32 out (unmasked)
__global__ void k_occ(const _Float16* __restrict__ A, const float* __restrict__ w,
                      const float* __restrict__ bb, float* __restrict__ out, int B)
{
    int i = blockIdx.x * blockDim.x + threadIdx.x;
    const int spat = DV * HV * WVX;
    if (i >= B * spat) return;
    int b = i / spat, r = i - b * spat;
    int d = r / (HV * WVX); r -= d * (HV * WVX);
    int h = r / WVX, v = r - (r / WVX) * WVX;
    float acc = bb[0];
    for (int kd = 0; kd < 3; ++kd) {
        int id = d + kd - 1; if ((unsigned)id >= (unsigned)DV) continue;
        for (int kh = 0; kh < 3; ++kh) {
            int ih = h + kh - 1; if ((unsigned)ih >= (unsigned)HV) continue;
            for (int kw = 0; kw < 3; ++kw) {
                int iw = v + kw - 1; if ((unsigned)iw >= (unsigned)WVX) continue;
                long pin = (((long)b * DV + id) * HV + ih) * (long)WVX + iw;
                int tap = (kd * 3 + kh) * 3 + kw;
                const _Float16* a = A + pin * 16;
                #pragma unroll
                for (int c = 0; c < 16; ++c) acc += w[c * 27 + tap] * (float)a[c];
            }
        }
    }
    out[i] = acc;
}

// fc1: per-voxel 16->3, * mask, (B,3,D,H,W) layout
__global__ void k_fc1(const _Float16* __restrict__ A, const float* __restrict__ w,
                      const float* __restrict__ bb, const float* __restrict__ mask,
                      float* __restrict__ out, int B)
{
    int i = blockIdx.x * blockDim.x + threadIdx.x;
    const int spat = DV * HV * WVX;
    if (i >= B * spat) return;
    int b = i / spat, v = i - b * spat;
    float x[16];
    #pragma unroll
    for (int c = 0; c < 16; ++c) x[c] = (float)A[(long)i * 16 + c];
    float mk = mask[i];
    #pragma unroll
    for (int o = 0; o < 3; ++o) {
        float z = bb[o];
        #pragma unroll
        for (int c = 0; c < 16; ++c) z += w[o * 16 + c] * x[c];
        out[((long)b * 3 + o) * spat + v] = z * mk;
    }
}

// ---------------------------------------------------------------------------
// Host-side orchestration
// ---------------------------------------------------------------------------
extern "C" void kernel_launch(void* const* d_in, const int* in_sizes, int n_in,
                              void* d_out, int out_size, void* d_ws, size_t ws_size,
                              hipStream_t stream)
{
    (void)out_size; (void)ws_size;
    const int B = 2;
    const int N = in_sizes[0] / (B * 9);

    // ---- input mapping (setup_inputs insertion order; params nested w,b,s,t) ----
    int idx = 0;
    const float* vfeat = (const float*)d_in[idx++];
    struct P4 { const float *w, *b, *s, *t; };
    struct P2 { const float *w, *b; };
    auto take4 = [&]() { P4 p; p.w = (const float*)d_in[idx++]; p.b = (const float*)d_in[idx++];
                         p.s = (const float*)d_in[idx++]; p.t = (const float*)d_in[idx++]; return p; };
    auto take2 = [&]() { P2 p; p.w = (const float*)d_in[idx++]; p.b = (const float*)d_in[idx++]; return p; };
    P4 e1 = take4();
    P4 e2d = take4(), e2s = take4();
    P4 e3d = take4(), e3s = take4();
    P4 e4d = take4(), e4s = take4();
    P4 e5d = take4(), e5s = take4();
    P4 d5u = take4(), d5s = take4(); P2 c5 = take2();
    P4 d4u = take4(), d4s = take4(); P2 c4 = take2();
    P4 d3u = take4(), d3s = take4(); P2 c3 = take2();
    P4 d2u = take4(), d2s = take4(); P2 c2 = take2();
    P2 occ = take2();
    P2 fc1 = take2();
    const int* coords = (const int*)d_in[n_in - 1];

    // ---- level geometry ----
    const int LD[5] = {96, 47, 23, 11, 5};
    const int LH[5] = {96, 47, 23, 11, 5};
    const int LW[5] = {48, 23, 11, 5, 2};
    int npos[5];
    for (int l = 0; l < 5; ++l) npos[l] = B * LD[l] * LH[l] * LW[l];

    // ---- workspace bump allocator (256B aligned) ----
    char* base = (char*)d_ws;
    size_t off = 0;
    auto alloc = [&](size_t bytes) -> void* {
        void* p = base + off;
        off += (bytes + 255) & ~(size_t)255;
        return p;
    };

    // activations (f16, channels-last)
    _Float16* Xin = (_Float16*)alloc((size_t)npos[0] * 16 * 2);
    _Float16* x0  = (_Float16*)alloc((size_t)npos[0] * 16 * 2);
    _Float16* u0  = (_Float16*)alloc((size_t)npos[0] * 16 * 2);
    _Float16* dd0 = (_Float16*)alloc((size_t)npos[0] * 16 * 2);
    _Float16* t1  = (_Float16*)alloc((size_t)npos[1] * 32 * 2);
    _Float16* x1  = (_Float16*)alloc((size_t)npos[1] * 32 * 2);
    _Float16* u1  = (_Float16*)alloc((size_t)npos[1] * 32 * 2);
    _Float16* dd1 = (_Float16*)alloc((size_t)npos[1] * 32 * 2);
    _Float16* t2  = (_Float16*)alloc((size_t)npos[2] * 64 * 2);
    _Float16* x2  = (_Float16*)alloc((size_t)npos[2] * 64 * 2);
    _Float16* u2  = (_Float16*)alloc((size_t)npos[2] * 64 * 2);
    _Float16* dd2 = (_Float16*)alloc((size_t)npos[2] * 64 * 2);
    _Float16* t3  = (_Float16*)alloc((size_t)npos[3] * 128 * 2);
    _Float16* x3  = (_Float16*)alloc((size_t)npos[3] * 128 * 2);
    _Float16* u3  = (_Float16*)alloc((size_t)npos[3] * 128 * 2);
    _Float16* dd3 = (_Float16*)alloc((size_t)npos[3] * 128 * 2);
    _Float16* t4  = (_Float16*)alloc((size_t)npos[4] * 256 * 2);
    _Float16* x4  = (_Float16*)alloc((size_t)npos[4] * 256 * 2);
    float* m[5];
    for (int l = 0; l < 5; ++l) m[l] = (float*)alloc((size_t)npos[l] * 4);

    // packed weights + folded affine
    struct LW_ { _Float16* Wp; float* sc; float* sh; int Ci, Co, Cipad, Kpad; };
    auto prep = [&](const P4& p, int Ci, int Co) -> LW_ {
        LW_ L;
        L.Ci = Ci; L.Co = Co;
        L.Cipad = (Ci + 15) & ~15;
        int Ktot = 27 * L.Cipad;
        L.Kpad = (Ktot + 31) & ~31;
        L.Wp = (_Float16*)alloc((size_t)Co * L.Kpad * 2);
        L.sc = (float*)alloc((size_t)Co * 4);
        L.sh = (float*)alloc((size_t)Co * 4);
        int tot = Co * L.Kpad;
        k_pack_weights<<<(tot + 255) / 256, 256, 0, stream>>>(
            p.w, p.b, p.s, p.t, L.Wp, L.sc, L.sh, Co, Ci, L.Cipad, L.Kpad);
        return L;
    };
    LW_ We1  = prep(e1,  9,   16);
    LW_ We2d = prep(e2d, 16,  32), We2s = prep(e2s, 32,  32);
    LW_ We3d = prep(e3d, 32,  64), We3s = prep(e3s, 64,  64);
    LW_ We4d = prep(e4d, 64, 128), We4s = prep(e4s, 128, 128);
    LW_ We5d = prep(e5d, 128,256), We5s = prep(e5s, 256, 256);
    LW_ Wd5u = prep(d5u, 256,128), Wd5s = prep(d5s, 128, 128);
    LW_ Wd4u = prep(d4u, 128, 64), Wd4s = prep(d4s, 64,  64);
    LW_ Wd3u = prep(d3u, 64,  32), Wd3s = prep(d3s, 32,  32);
    LW_ Wd2u = prep(d2u, 32,  16), Wd2s = prep(d2s, 16,  16);

    // ---- densify ----
    hipMemsetAsync(Xin, 0, (size_t)npos[0] * 16 * 2, stream);
    hipMemsetAsync(m[0], 0, (size_t)npos[0] * 4, stream);
    k_densify<<<(B * N + 255) / 256, 256, 0, stream>>>(vfeat, coords, Xin, m[0], B, N);

    // ---- mask pyramid ----
    for (int l = 0; l < 4; ++l)
        k_down_mask<<<(npos[l + 1] + 255) / 256, 256, 0, stream>>>(
            m[l], m[l + 1], B, LD[l], LH[l], LW[l], LD[l + 1], LH[l + 1], LW[l + 1]);

    // ---- generic conv launcher ----
    auto conv = [&](const _Float16* Xa, const LW_& L, const float* mk,
                    _Float16* Ya, int li, int lo, int mode) {
        int NT = (L.Co == 16) ? 1 : (L.Co == 32) ? 2 : 4;
        run_conv(L.Cipad, mode, NT, Xa, L.Wp, L.sc, L.sh, mk, Ya, B,
                 LD[li], LH[li], LW[li], LD[lo], LH[lo], LW[lo], L.Co, stream);
    };

    // ---- encoder ----
    conv(Xin, We1,  m[0], x0, 0, 0, 0);
    conv(x0,  We2d, m[1], t1, 0, 1, 1);  conv(t1, We2s, m[1], x1, 1, 1, 0);
    conv(x1,  We3d, m[2], t2, 1, 2, 1);  conv(t2, We3s, m[2], x2, 2, 2, 0);
    conv(x2,  We4d, m[3], t3, 2, 3, 1);  conv(t3, We4s, m[3], x3, 3, 3, 0);
    conv(x3,  We5d, m[4], t4, 3, 4, 1);  conv(t4, We5s, m[4], x4, 4, 4, 0);

    // ---- output segment offsets (return order) ----
    float* out = (float*)d_out;
    const long spat0 = (long)LD[0] * LH[0] * LW[0];
    float* o_coords = out;
    float* o_occ    = o_coords + (long)B * 3 * spat0;
    float* o_p5     = o_occ + (long)B * spat0;
    float* o_p4     = o_p5 + npos[3];
    float* o_p3     = o_p4 + npos[2];
    float* o_p2     = o_p3 + npos[1];

    // ---- decoder + heads ----
    conv(x4, Wd5u, m[3], u3, 4, 3, 2);   conv(u3, Wd5s, m[3], dd3, 3, 3, 0);
    k_cls<<<(npos[3] + 255) / 256, 256, 0, stream>>>(dd3, c5.w, c5.b, m[3], o_p5, npos[3], 128, 128);
    k_add<<<(int)(((long)npos[3] * 128 + 255) / 256), 256, 0, stream>>>(dd3, x3, (long)npos[3] * 128);

    conv(dd3, Wd4u, m[2], u2, 3, 2, 2);  conv(u2, Wd4s, m[2], dd2, 2, 2, 0);
    k_cls<<<(npos[2] + 255) / 256, 256, 0, stream>>>(dd2, c4.w, c4.b, m[2], o_p4, npos[2], 64, 64);
    k_add<<<(int)(((long)npos[2] * 64 + 255) / 256), 256, 0, stream>>>(dd2, x2, (long)npos[2] * 64);

    conv(dd2, Wd3u, m[1], u1, 2, 1, 2);  conv(u1, Wd3s, m[1], dd1, 1, 1, 0);
    k_cls<<<(npos[1] + 255) / 256, 256, 0, stream>>>(dd1, c3.w, c3.b, m[1], o_p3, npos[1], 32, 32);
    k_add<<<(int)(((long)npos[1] * 32 + 255) / 256), 256, 0, stream>>>(dd1, x1, (long)npos[1] * 32);

    conv(dd1, Wd2u, m[0], u0, 1, 0, 2);  conv(u0, Wd2s, m[0], dd0, 0, 0, 0);
    k_cls<<<(npos[0] + 255) / 256, 256, 0, stream>>>(dd0, c2.w, c2.b, m[0], o_p2, npos[0], 16, 16);

    k_occ<<<(npos[0] + 255) / 256, 256, 0, stream>>>(dd0, occ.w, occ.b, o_occ, B);
    k_fc1<<<(npos[0] + 255) / 256, 256, 0, stream>>>(dd0, fc1.w, fc1.b, m[0], o_coords, B);
}
